// MMDLoss_59356448031516
// MI455X (gfx1250) — compile-verified, matrix operand order checked by reading, not run on hardware
//
#include <hip/hip_runtime.h>

typedef __attribute__((ext_vector_type(16))) _Float16 v16h;
typedef __attribute__((ext_vector_type(8)))  _Float16 v8h;
typedef __attribute__((ext_vector_type(8)))  float    v8f;

#define DIMS 128
#define LOG2E 1.4426950408889634f

// Convert one row f32 -> f16 (one wave per row) and compute the squared norm
// of the *converted* values so that dist(i,i) == 0 in the WMMA path.
__global__ __launch_bounds__(256) void cvt_norm_kernel(
    const float* __restrict__ src, _Float16* __restrict__ dst,
    float* __restrict__ nrm, int rows)
{
    const int row  = blockIdx.x * 8 + (threadIdx.x >> 5);
    const int lane = threadIdx.x & 31;
    if (row >= rows) return;
    const float* s = src + (size_t)row * DIMS;
    _Float16*    d = dst + (size_t)row * DIMS;
    float sq = 0.f;
#pragma unroll
    for (int t = 0; t < DIMS / 32; ++t) {
        float    v = s[lane + t * 32];
        _Float16 h = (_Float16)v;
        d[lane + t * 32] = h;
        float hv = (float)h;
        sq += hv * hv;
    }
#pragma unroll
    for (int off = 16; off > 0; off >>= 1) sq += __shfl_xor(sq, off, 32);
    if (lane == 0) nrm[row] = sq;
}

__global__ void zero_kernel(float* out) { out[0] = 0.f; }

// One block (16 waves) per 32-row i-block of A. Each wave keeps the 32x128 A
// block resident as 2x4 v16h fragments and strides over 16-col j-tiles of B
// with double-buffered fragment prefetch (unroll-2 ping-pong, no reg copies).
// Per B tile: 8 WMMAs across two independent accumulator chains.
// Wave-uniform control (tile index, weights, addresses) is scalarized via
// readfirstlane so it runs on the SALU and co-issues with VALU/WMMA.
// SYM=1: A==B, sub-tile weights {0,1,2} implement triangle counting.
template <int SYM>
__global__ __launch_bounds__(512) void mmd_pair_kernel(
    const _Float16* __restrict__ A, const _Float16* __restrict__ B,
    const float* __restrict__ nA, const float* __restrict__ nB,
    float w, float* __restrict__ out, int N)
{
    const int wave = __builtin_amdgcn_readfirstlane(threadIdx.x >> 5); // SGPR
    const int lane = threadIdx.x & 31;
    const int ib   = blockIdx.x;         // 32-row block index
    const int it0  = ib << 1;            // first 16-row tile index
    const int i0   = ib << 5;
    const int m    = lane & 15;
    const int hi   = lane >> 4;
    const int loff = m * DIMS + hi * 8;  // loop-invariant per-lane elem offset

    union V16 { v16h v; v8h h[2]; };

    // Resident A fragments: two 16-row tiles x four K=32 chunks.
    V16 af[2][4];
#pragma unroll
    for (int s = 0; s < 2; ++s) {
        const _Float16* p = A + (size_t)(i0 + s * 16) * DIMS + loff;
#pragma unroll
        for (int kc = 0; kc < 4; ++kc) {
            af[s][kc].h[0] = *(const v8h*)(p + kc * 32);
            af[s][kc].h[1] = *(const v8h*)(p + kc * 32 + 16);
        }
    }

    // Row norms pre-scaled into the exp2 domain: sa = -log2(e) * ||a||^2.
    float sa[2][8];
#pragma unroll
    for (int s = 0; s < 2; ++s)
#pragma unroll
        for (int r = 0; r < 8; ++r)
            sa[s][r] = -LOG2E * nA[i0 + s * 16 + r + hi * 8];

    const int nTiles = N >> 4;
    // First j-tile for this wave: stride-16 class, >= it0 when symmetric.
    int jt = SYM ? (it0 + ((wave - it0) & 15)) : wave;

    float acc = 0.f;
    if (jt < nTiles) {
        V16 bc[4];
        float nbc;
        {
            const _Float16* p = B + (size_t)(jt << 4) * DIMS + loff;
#pragma unroll
            for (int kc = 0; kc < 4; ++kc) {
                bc[kc].h[0] = *(const v8h*)(p + kc * 32);
                bc[kc].h[1] = *(const v8h*)(p + kc * 32 + 16);
            }
            nbc = nB[(jt << 4) + m];
        }
#pragma unroll 2
        for (; jt < nTiles; jt += 16) {
            // Prefetch next tile (clamped; scalar address math).
            const int jn = (jt + 16 < nTiles) ? jt + 16 : jt;
            V16 bn[4];
            float nbn;
            {
                const _Float16* p = B + (size_t)(jn << 4) * DIMS + loff;
#pragma unroll
                for (int kc = 0; kc < 4; ++kc) {
                    bn[kc].h[0] = *(const v8h*)(p + kc * 32);
                    bn[kc].h[1] = *(const v8h*)(p + kc * 32 + 16);
                }
                nbn = nB[(jn << 4) + m];
            }

            // 8 WMMAs over the current buffer, two independent chains.
            v8f c0 = {}, c1 = {};
#pragma unroll
            for (int kc = 0; kc < 4; ++kc) {
                c0 = __builtin_amdgcn_wmma_f32_16x16x32_f16(
                         false, af[0][kc].v, false, bc[kc].v, (short)0, c0, false, false);
                c1 = __builtin_amdgcn_wmma_f32_16x16x32_f16(
                         false, af[1][kc].v, false, bc[kc].v, (short)0, c1, false, false);
            }

            // exp(-(na+nb-2c)) = exp2(sa + sb + (2*log2e)*c), one fma+exp2 each.
            const float sb = -LOG2E * nbc;
            float e0[8], e1[8];
#pragma unroll
            for (int r = 0; r < 8; ++r) {
                e0[r] = __builtin_amdgcn_exp2f(
                            __builtin_fmaf(c0[r], 2.f * LOG2E, sa[0][r] + sb));
                e1[r] = __builtin_amdgcn_exp2f(
                            __builtin_fmaf(c1[r], 2.f * LOG2E, sa[1][r] + sb));
            }
            float ts0 = ((e0[0] + e0[1]) + (e0[2] + e0[3])) +
                        ((e0[4] + e0[5]) + (e0[6] + e0[7]));
            float ts1 = ((e1[0] + e1[1]) + (e1[2] + e1[3])) +
                        ((e1[4] + e1[5]) + (e1[6] + e1[7]));

            if (SYM) {
                // Scalar (SGPR) weight selection: {0,1,2} per sub-tile.
                const float w0 = (jt == it0)     ? 1.f : (jt > it0     ? 2.f : 0.f);
                const float w1 = (jt == it0 + 1) ? 1.f : (jt > it0 + 1 ? 2.f : 0.f);
                acc = __builtin_fmaf(w0, ts0, acc);
                acc = __builtin_fmaf(w1, ts1, acc);
            } else {
                acc += ts0 + ts1;
            }

#pragma unroll
            for (int kc = 0; kc < 4; ++kc) bc[kc] = bn[kc];
            nbc = nbn;
        }
    }

    acc *= w;
#pragma unroll
    for (int off = 16; off > 0; off >>= 1) acc += __shfl_xor(acc, off, 32);
    if (lane == 0) atomicAdd(out, acc);
}

extern "C" void kernel_launch(void* const* d_in, const int* in_sizes, int n_in,
                              void* d_out, int out_size, void* d_ws, size_t ws_size,
                              hipStream_t stream) {
    const float* x = (const float*)d_in[0];
    const float* y = (const float*)d_in[1];
    float* out = (float*)d_out;
    const int N = in_sizes[0] / DIMS;   // 8192

    char* ws = (char*)d_ws;
    const size_t matBytes = (size_t)N * DIMS * sizeof(_Float16);
    _Float16* xh = (_Float16*)ws;
    _Float16* yh = (_Float16*)(ws + matBytes);
    float*    nx = (float*)(ws + 2 * matBytes);
    float*    ny = nx + N;

    const int rowBlocks = (N + 7) / 8;
    cvt_norm_kernel<<<rowBlocks, 256, 0, stream>>>(x, xh, nx, N);
    cvt_norm_kernel<<<rowBlocks, 256, 0, stream>>>(y, yh, ny, N);
    zero_kernel<<<1, 1, 0, stream>>>(out);

    const float invN2 = 1.f / ((float)N * (float)N);
    mmd_pair_kernel<1><<<N / 32, 512, 0, stream>>>(xh, xh, nx, nx,        invN2, out, N);
    mmd_pair_kernel<1><<<N / 32, 512, 0, stream>>>(yh, yh, ny, ny,        invN2, out, N);
    mmd_pair_kernel<0><<<N / 32, 512, 0, stream>>>(xh, yh, nx, ny, -2.f * invN2, out, N);
}
